// CausalSelfAttention_11115375362048
// MI455X (gfx1250) — matrix operand order repaired and checked by reading, not compiled, not run on hardware
//
#include <hip/hip_runtime.h>
#include <hip/hip_bf16.h>

// ---------------------------------------------------------------------------
// Causal self-attention forward for MI455X (gfx1250, wave32, WMMA).
// Stages: [pre-convert x, W^T to bf16] -> [QKV GEMM, V stored transposed]
//         -> [flash attention] -> [proj GEMM].
// All matmuls on v_wmma_f32_16x16x32_bf16; GEMMs and the attention key loop
// are software-pipelined (register double-buffered fragments).
// ---------------------------------------------------------------------------

typedef __attribute__((ext_vector_type(16))) __bf16 v16bf;
typedef __attribute__((ext_vector_type(8)))  float  v8f;

constexpr int kB = 2, kT = 2048, kC = 1024, kH = 16, kD = 64;
constexpr int kM = kB * kT;          // 4096 rows
constexpr int kN1 = 3 * kC;          // 3072 qkv columns

__device__ __forceinline__ unsigned short f2bf(float f) {
  return __builtin_bit_cast(unsigned short, (__bf16)f);
}

__device__ __forceinline__ v8f wmma_bf16(v16bf a, v16bf b, v8f c) {
  return __builtin_amdgcn_wmma_f32_16x16x32_bf16(false, a, false, b,
                                                 (short)0, c, false, false);
}

// A-fragment loader (contiguous K order, matching the B loads):
// lanes 0-15: K = {0..15}; lanes 16-31: K = {16..31}.
struct APair { uint4 lo, hi; };
__device__ __forceinline__ v16bf load_a_frag(const unsigned short* p, int ca) {
  APair u;
  u.lo = *(const uint4*)(p + 2 * ca);
  u.hi = *(const uint4*)(p + 2 * ca + 8);
  return __builtin_bit_cast(v16bf, u);
}

__device__ __forceinline__ float redmax16(float v) {
  v = fmaxf(v, __shfl_xor(v, 1, 16));
  v = fmaxf(v, __shfl_xor(v, 2, 16));
  v = fmaxf(v, __shfl_xor(v, 4, 16));
  v = fmaxf(v, __shfl_xor(v, 8, 16));
  return v;
}
__device__ __forceinline__ float redsum16(float v) {
  v += __shfl_xor(v, 1, 16);
  v += __shfl_xor(v, 2, 16);
  v += __shfl_xor(v, 4, 16);
  v += __shfl_xor(v, 8, 16);
  return v;
}

// Software-pipelined 16x64 GEMM tile over K=kC: preload next k-step's
// fragments, then issue 4 WMMAs on the current step; last step peeled.
__device__ __forceinline__ void gemm_tile_16x64(
    const unsigned short* __restrict__ arow,   // A row base (+lane row)
    const unsigned short* __restrict__ bbase,  // W^T row for f=0 column (+kgB)
    int ca, v8f (&acc)[4])
{
  v16bf a  = load_a_frag(arow, ca);
  v16bf b0 = *(const v16bf*)(bbase);
  v16bf b1 = *(const v16bf*)(bbase + 16 * kC);
  v16bf b2 = *(const v16bf*)(bbase + 32 * kC);
  v16bf b3 = *(const v16bf*)(bbase + 48 * kC);
  for (int k0 = 0; k0 < kC - 32; k0 += 32) {
    const int kn = k0 + 32;
    const v16bf an  = load_a_frag(arow + kn, ca);
    const v16bf nb0 = *(const v16bf*)(bbase + kn);
    const v16bf nb1 = *(const v16bf*)(bbase + 16 * kC + kn);
    const v16bf nb2 = *(const v16bf*)(bbase + 32 * kC + kn);
    const v16bf nb3 = *(const v16bf*)(bbase + 48 * kC + kn);
    acc[0] = wmma_bf16(a, b0, acc[0]);
    acc[1] = wmma_bf16(a, b1, acc[1]);
    acc[2] = wmma_bf16(a, b2, acc[2]);
    acc[3] = wmma_bf16(a, b3, acc[3]);
    a = an; b0 = nb0; b1 = nb1; b2 = nb2; b3 = nb3;
  }
  acc[0] = wmma_bf16(a, b0, acc[0]);
  acc[1] = wmma_bf16(a, b1, acc[1]);
  acc[2] = wmma_bf16(a, b2, acc[2]);
  acc[3] = wmma_bf16(a, b3, acc[3]);
}

// ---------------------------------------------------------------------------
// Pre-conversion kernels (one-shot, L2-resident data).
// ---------------------------------------------------------------------------
__global__ __launch_bounds__(256) void cvt_bf16_kernel(
    const float* __restrict__ s, unsigned short* __restrict__ d, long n)
{
  const long i = ((long)blockIdx.x * blockDim.x + threadIdx.x) * 8;
  if (i + 8 > n) return;
  const float4 f0 = ((const float4*)(s + i))[0];
  const float4 f1 = ((const float4*)(s + i))[1];
  uint4 u;
  u.x = (unsigned)f2bf(f0.x) | ((unsigned)f2bf(f0.y) << 16);
  u.y = (unsigned)f2bf(f0.z) | ((unsigned)f2bf(f0.w) << 16);
  u.z = (unsigned)f2bf(f1.x) | ((unsigned)f2bf(f1.y) << 16);
  u.w = (unsigned)f2bf(f1.z) | ((unsigned)f2bf(f1.w) << 16);
  *(uint4*)(d + i) = u;
}

// dst[n*K + k] = bf16(src[k*N + n]) : weights K x N (row-major) -> N x K.
__global__ __launch_bounds__(256) void transpose_bf16_kernel(
    const float* __restrict__ src, unsigned short* __restrict__ dst,
    int K, int N)
{
  const long idx = (long)blockIdx.x * blockDim.x + threadIdx.x;
  const long n4 = N >> 2;
  const int k = (int)(idx / n4);
  const int n = (int)(idx % n4) << 2;
  if (k >= K) return;
  const float4 f = *(const float4*)(src + (size_t)k * N + n);
  dst[(size_t)(n + 0) * K + k] = f2bf(f.x);
  dst[(size_t)(n + 1) * K + k] = f2bf(f.y);
  dst[(size_t)(n + 2) * K + k] = f2bf(f.z);
  dst[(size_t)(n + 3) * K + k] = f2bf(f.w);
}

// ---------------------------------------------------------------------------
// QKV GEMM: epilogue routes to Q (pre-scaled 1/sqrt(D)), K, or transposed V.
// ---------------------------------------------------------------------------
__global__ __launch_bounds__(256) void gemm_qkv_kernel(
    const unsigned short* __restrict__ xbf,
    const unsigned short* __restrict__ wt,   // [3C][C] bf16 (W^T)
    const float* __restrict__ bias,
    unsigned short* __restrict__ Qb, unsigned short* __restrict__ Kb,
    unsigned short* __restrict__ Vt)
{
  const int lane = threadIdx.x & 31;
  const int wave = threadIdx.x >> 5;
  const int ntiles = kN1 / 64;                       // 48
  const long tile = (long)blockIdx.x * 8 + wave;
  if (tile >= (long)(kM / 16) * ntiles) return;      // wave-uniform
  const int nt = (int)(tile % ntiles);
  const int mt = (int)(tile / ntiles);

  const int hiK = lane >> 4;
  const int ca  = hiK ? 8 : 0;
  const int mA  = lane & 15;
  const int nB  = lane & 15;
  const int kgB = hiK * 16;

  v8f acc[4] = {{}, {}, {}, {}};
  gemm_tile_16x64(xbf + (size_t)(mt * 16 + mA) * kC,
                  wt + (size_t)(nt * 64 + nB) * kC + kgB, ca, acc);

  const int hiC = hiK ? 8 : 0;
  const int nC  = lane & 15;
  const int colbase = nt * 64;
  const int part = colbase >> 10;                    // wave-uniform
  const int h    = (colbase & (kC - 1)) >> 6;        // wave-uniform
#pragma unroll
  for (int f = 0; f < 4; ++f) {
    const int d = f * 16 + nC;
    const float bv = bias[colbase + d];
#pragma unroll
    for (int r = 0; r < 8; ++r) {
      const int row = mt * 16 + hiC + r;
      const float v = acc[f][r] + bv;
      if (part == 0) {
        Qb[(size_t)row * kC + h * kD + d] = f2bf(v * 0.125f);
      } else if (part == 1) {
        Kb[(size_t)row * kC + h * kD + d] = f2bf(v);
      } else {
        const int bb = row >> 11, t = row & (kT - 1);
        Vt[((size_t)(bb * kH + h) * kD + d) * kT + t] = f2bf(v);
      }
    }
  }
}

// ---------------------------------------------------------------------------
// Proj GEMM: out = y @ Wproj + bproj (bf16 x bf16 -> f32).
// ---------------------------------------------------------------------------
__global__ __launch_bounds__(256) void gemm_proj_kernel(
    const unsigned short* __restrict__ ybf,
    const unsigned short* __restrict__ wt,   // [C][C] bf16 (W^T)
    const float* __restrict__ bias, float* __restrict__ out)
{
  const int lane = threadIdx.x & 31;
  const int wave = threadIdx.x >> 5;
  const int ntiles = kC / 64;                        // 16
  const long tile = (long)blockIdx.x * 8 + wave;
  if (tile >= (long)(kM / 16) * ntiles) return;
  const int nt = (int)(tile % ntiles);
  const int mt = (int)(tile / ntiles);

  const int hiK = lane >> 4;
  const int ca  = hiK ? 8 : 0;
  const int mA  = lane & 15;
  const int kgB = hiK * 16;

  v8f acc[4] = {{}, {}, {}, {}};
  gemm_tile_16x64(ybf + (size_t)(mt * 16 + mA) * kC,
                  wt + (size_t)(nt * 64 + (lane & 15)) * kC + kgB, ca, acc);

  const int hiC = hiK ? 8 : 0;
  const int nC  = lane & 15;
#pragma unroll
  for (int f = 0; f < 4; ++f) {
    const int col = nt * 64 + f * 16 + nC;
    const float bv = bias[col];
#pragma unroll
    for (int r = 0; r < 8; ++r) {
      const int row = mt * 16 + hiC + r;
      out[(size_t)row * kC + col] = acc[f][r] + bv;
    }
  }
}

// ---------------------------------------------------------------------------
// Flash attention: one wave per (b, h, 16-row q tile), 32-key blocks.
// K fragments pipelined one block ahead; V fragments issued before the
// softmax so their latency hides under ~30 VALU ops; P transposed C->A
// layout via 1KB/wave LDS guarded by s_wait_dscnt.
// ---------------------------------------------------------------------------
__global__ __launch_bounds__(256) void attn_fwd_kernel(
    const unsigned short* __restrict__ Qb, const unsigned short* __restrict__ Kb,
    const unsigned short* __restrict__ Vt, unsigned short* __restrict__ y)
{
  __shared__ __align__(32) unsigned short Plds[8][16 * 32];

  const int lane = threadIdx.x & 31;
  const int wave = threadIdx.x >> 5;
  const int QT = kT / 16;
  const int w = blockIdx.x * 8 + wave;
  if (w >= kB * kH * QT) return;             // wave-uniform
  const int qt = w % QT;
  const int h  = (w / QT) % kH;
  const int b  = w / (QT * kH);

  const int hiK = lane >> 4;
  const int nC  = lane & 15;
  const int hiC = hiK ? 8 : 0;
  const int ca  = hiK ? 8 : 0;
  const int mA  = lane & 15;

  const unsigned short* Qrow  = Qb + ((size_t)b * kT + qt * 16 + mA) * kC + h * kD;
  const unsigned short* Kbase = Kb + (size_t)b * kT * kC + h * kD;
  const unsigned short* Vbase = Vt + (size_t)(b * kH + h) * kD * kT;

  v16bf qa[2];
#pragma unroll
  for (int kk = 0; kk < 2; ++kk) qa[kk] = load_a_frag(Qrow + kk * 32, ca);

  // K fragment loader: idx = 2*kk + half (kk = d-step, half = key half).
  auto loadK = [&](v16bf (&dst)[4], int kbv) {
    const unsigned short* r0 = Kbase + (size_t)(kbv + nC) * kC + hiK * 16;
    const unsigned short* r1 = Kbase + (size_t)(kbv + 16 + nC) * kC + hiK * 16;
    dst[0] = *(const v16bf*)(r0);
    dst[1] = *(const v16bf*)(r1);
    dst[2] = *(const v16bf*)(r0 + 32);
    dst[3] = *(const v16bf*)(r1 + 32);
  };

  v8f o0 = {}, o1 = {}, o2 = {}, o3 = {};
  float rmax[8], rsum[8];
#pragma unroll
  for (int r = 0; r < 8; ++r) { rmax[r] = -1e30f; rsum[r] = 0.f; }

  const int   qhi = qt * 16 + 15;
  const float L2E = 1.4426950408889634f;

  v16bf kcur[4];
  loadK(kcur, 0);

  for (int kb = 0; kb <= qhi; kb += 32) {
    // Issue V loads for this block and K loads for the next block first:
    // both complete under the softmax below.
    v16bf vb[4];
#pragma unroll
    for (int f = 0; f < 4; ++f)
      vb[f] = *(const v16bf*)(Vbase + (size_t)(f * 16 + nC) * kT + kb + hiK * 16);

    const int kbn = (kb + 32 <= qhi) ? (kb + 32) : kb;  // uniform clamp
    v16bf knx[4];
    loadK(knx, kbn);

    if (kb + 64 <= qhi) {  // L2 prefetch two blocks ahead (uniform guard)
      __builtin_prefetch(Kbase + (size_t)(kb + 64 + nC) * kC, 0, 1);
      __builtin_prefetch(Vbase + (size_t)nC * kT + kb + 64, 0, 1);
    }

    // ---- S = Q K^T (Q pre-scaled by 1/sqrt(D)) ----
    v8f s0 = {}, s1 = {};
#pragma unroll
    for (int kk = 0; kk < 2; ++kk) {
      s0 = wmma_bf16(qa[kk], kcur[kk * 2 + 0], s0);
      s1 = wmma_bf16(qa[kk], kcur[kk * 2 + 1], s1);
    }

    // ---- online softmax ----
    const bool needMask = (kb + 31) > qt * 16;
    float p0[8], p1[8], alpha[8];
#pragma unroll
    for (int r = 0; r < 8; ++r) {
      float a0 = s0[r], a1 = s1[r];
      if (needMask) {
        const int q = qt * 16 + hiC + r;
        a0 = (kb + nC > q)      ? -1e30f : a0;
        a1 = (kb + 16 + nC > q) ? -1e30f : a1;
      }
      const float mloc = redmax16(fmaxf(a0, a1));
      const float mnew = fmaxf(rmax[r], mloc);
      alpha[r] = exp2f((rmax[r] - mnew) * L2E);
      rmax[r]  = mnew;
      p0[r] = exp2f((a0 - mnew) * L2E);
      p1[r] = exp2f((a1 - mnew) * L2E);
      rsum[r] = rsum[r] * alpha[r] + redsum16(p0[r] + p1[r]);
    }

#pragma unroll
    for (int r = 0; r < 8; ++r) {
      o0[r] *= alpha[r]; o1[r] *= alpha[r]; o2[r] *= alpha[r]; o3[r] *= alpha[r];
      Plds[wave][(hiC + r) * 32 + nC]      = f2bf(p0[r]);
      Plds[wave][(hiC + r) * 32 + 16 + nC] = f2bf(p1[r]);
    }
    asm volatile("s_wait_dscnt 0x0" ::: "memory");

    const v16bf pa = load_a_frag(&Plds[wave][mA * 32], ca);

    // ---- O += P V (V fragments already resident) ----
    o0 = wmma_bf16(pa, vb[0], o0);
    o1 = wmma_bf16(pa, vb[1], o1);
    o2 = wmma_bf16(pa, vb[2], o2);
    o3 = wmma_bf16(pa, vb[3], o3);

#pragma unroll
    for (int i = 0; i < 4; ++i) kcur[i] = knx[i];
  }

#pragma unroll
  for (int r = 0; r < 8; ++r) {
    const float inv = 1.f / rsum[r];
    const int t = qt * 16 + hiC + r;
    unsigned short* yr = y + ((size_t)b * kT + t) * kC + (size_t)h * kD;
    yr[0 * 16 + nC] = f2bf(o0[r] * inv);
    yr[1 * 16 + nC] = f2bf(o1[r] * inv);
    yr[2 * 16 + nC] = f2bf(o2[r] * inv);
    yr[3 * 16 + nC] = f2bf(o3[r] * inv);
  }
}

// ---------------------------------------------------------------------------
extern "C" void kernel_launch(void* const* d_in, const int* in_sizes, int n_in,
                              void* d_out, int out_size, void* d_ws, size_t ws_size,
                              hipStream_t stream) {
  const float* x     = (const float*)d_in[0];
  const float* Wqkv  = (const float*)d_in[1];
  const float* bqkv  = (const float*)d_in[2];
  const float* Wproj = (const float*)d_in[3];
  const float* bproj = (const float*)d_in[4];
  float* out = (float*)d_out;

  // workspace layout (bf16 elements)
  unsigned short* p = (unsigned short*)d_ws;
  unsigned short* xbf    = p; p += (size_t)kM * kC;
  unsigned short* wqkvT  = p; p += (size_t)kN1 * kC;
  unsigned short* wprojT = p; p += (size_t)kC * kC;
  unsigned short* Qb     = p; p += (size_t)kM * kC;
  unsigned short* Kb     = p; p += (size_t)kM * kC;
  unsigned short* Vt     = p; p += (size_t)kM * kC;
  unsigned short* ybf    = p;

  {
    const long n = (long)kM * kC;
    cvt_bf16_kernel<<<(int)(n / 8 / 256), 256, 0, stream>>>(x, xbf, n);
  }
  transpose_bf16_kernel<<<(int)((long)kC * kN1 / 4 / 256), 256, 0, stream>>>(
      Wqkv, wqkvT, kC, kN1);
  transpose_bf16_kernel<<<(int)((long)kC * kC / 4 / 256), 256, 0, stream>>>(
      Wproj, wprojT, kC, kC);

  {
    const long tiles = (long)(kM / 16) * (kN1 / 64);
    gemm_qkv_kernel<<<(int)((tiles + 7) / 8), 256, 0, stream>>>(
        xbf, wqkvT, bqkv, Qb, Kb, Vt);
  }
  {
    const int waves = kB * kH * (kT / 16);
    attn_fwd_kernel<<<(waves + 7) / 8, 256, 0, stream>>>(Qb, Kb, Vt, ybf);
  }
  {
    const long tiles = (long)(kM / 16) * (kC / 64);
    gemm_proj_kernel<<<(int)((tiles + 7) / 8), 256, 0, stream>>>(
        ybf, wprojT, bproj, out);
  }
}
